// RNNVAENET_17918603558926
// MI455X (gfx1250) — compile-verified, matrix operand order checked by reading, not run on hardware
//
#include <hip/hip_runtime.h>

// ---------------------------------------------------------------------------
// Types for WMMA
// ---------------------------------------------------------------------------
typedef __bf16  v16bf __attribute__((ext_vector_type(16)));
typedef float   v8f   __attribute__((ext_vector_type(8)));
typedef unsigned int uint4v __attribute__((ext_vector_type(4)));

union Frag {
    v16bf bf;
    uint4v u[2];
};

// float -> bf16 bits, round-to-nearest-even
__device__ __forceinline__ unsigned short f2bf(float f) {
    unsigned int u = __float_as_uint(f);
    unsigned int r = (u + 0x7FFFu + ((u >> 16) & 1u)) >> 16;
    return (unsigned short)r;
}

// ---------------------------------------------------------------------------
// Problem constants
// ---------------------------------------------------------------------------
#define BB   2048
#define TT   64
#define NROWS (BB * TT)       // 131072
#define LAT  64
#define HH   8

// Block geometry for the WMMA MLP kernels
#define MROWS 64              // rows per block (4 M-tiles)
#define NWAVE 4               // waves per block (split N-tiles)

// ---------------------------------------------------------------------------
// Weight prep: W (din x dout, row-major fp32) -> Wt (dout_pad x din_pad bf16),
// bias -> padded fp32
// ---------------------------------------------------------------------------
__global__ void prep_weight(const float* __restrict__ W, const float* __restrict__ b,
                            unsigned short* __restrict__ Wt, float* __restrict__ bp,
                            int din, int dout, int din_pad, int dout_pad) {
    int idx = blockIdx.x * blockDim.x + threadIdx.x;
    int total = din_pad * dout_pad;
    if (idx >= total) return;
    int n = idx / din_pad;
    int k = idx - n * din_pad;
    float v = (n < dout && k < din) ? W[(long)k * dout + n] : 0.f;
    Wt[idx] = f2bf(v);
    if (k == 0) bp[n] = (n < dout) ? b[n] : 0.f;
}

// ---------------------------------------------------------------------------
// Build padded bf16 encoder input: rows = B*T, 64 cols (27 lower + 18 hands)
// ---------------------------------------------------------------------------
__global__ void build_enc_input(const float* __restrict__ lb, const float* __restrict__ hands,
                                unsigned short* __restrict__ X) {
    long idx = (long)blockIdx.x * blockDim.x + threadIdx.x;
    const long total = (long)NROWS * 64;
    if (idx >= total) return;
    int col = (int)(idx & 63);
    long row = idx >> 6;
    int b = (int)(row >> 6);   // T == 64
    float v = 0.f;
    if (col < 27)      v = lb[row * 27 + col];
    else if (col < 45) v = hands[b * 18 + (col - 27)];
    X[idx] = f2bf(v);
}

// ---------------------------------------------------------------------------
// Build padded bf16 decoder input: 2048 rows x 32 cols (8 rnn_last + 18 hands)
// ---------------------------------------------------------------------------
__global__ void build_dec_input(const float* __restrict__ rnnLast, const float* __restrict__ hands,
                                unsigned short* __restrict__ X) {
    int idx = blockIdx.x * blockDim.x + threadIdx.x;
    if (idx >= BB * 32) return;
    int col = idx & 31;
    int b = idx >> 5;
    float v = 0.f;
    if (col < 8)       v = rnnLast[b * 8 + col];
    else if (col < 26) v = hands[b * 18 + (col - 8)];
    X[idx] = f2bf(v);
}

// ---------------------------------------------------------------------------
// Layer descriptor
// ---------------------------------------------------------------------------
struct Layer {
    const unsigned short* Wt;   // [dout_pad][din_pad] bf16, K contiguous
    const float* bias;          // [dout_pad]
    int din_pad;                // multiple of 32
    int dout_pad;               // multiple of 16
    int dout;                   // real output cols
    int act;                    // 1 -> leaky_relu(0.01)
};

// Shared tile-load helper: copy 64 x dp bf16 rows (contiguous X) into LDS
// (row stride 512) using uint (2-element) copies.
__device__ __forceinline__ void load_tile(const unsigned short* __restrict__ X,
                                          long row0, int dp,
                                          unsigned short* __restrict__ act, int tid) {
    const unsigned* src = (const unsigned*)(X + row0 * dp);
    const int half = (MROWS * dp) >> 1;
    for (int i = tid; i < half; i += 32 * NWAVE) {
        int e = i << 1;
        int r = e / dp, c = e - r * dp;
        *(unsigned*)(act + r * 512 + c) = src[i];
    }
}

// ---------------------------------------------------------------------------
// Core register-blocked GEMM step: one wave computes a 64-row x 32-col patch
// (2 N-tiles x 4 M-tiles, 8 accumulators).  B fragments reused 4x, A
// fragments reused 2x -> halved LDS traffic per WMMA vs 1-N-tile blocking.
// ---------------------------------------------------------------------------
__device__ __forceinline__ void gemm_patch(const unsigned short* __restrict__ cur,
                                           const unsigned short* __restrict__ wrow0,
                                           const unsigned short* __restrict__ wrow1,
                                           int kt_n, int hi, int ln, v8f acc[2][4]) {
#pragma unroll 2
    for (int kt = 0; kt < kt_n; ++kt) {
        const int kb = kt * 32 + hi * 8;
        Frag B0, B1;
        B0.u[0] = *(const uint4v*)(wrow0 + kb);
        B0.u[1] = *(const uint4v*)(wrow0 + kb + 16);
        B1.u[0] = *(const uint4v*)(wrow1 + kb);
        B1.u[1] = *(const uint4v*)(wrow1 + kb + 16);
        __builtin_prefetch(wrow0 + kb + 64, 0, 1);
        __builtin_prefetch(wrow1 + kb + 64, 0, 1);
#pragma unroll
        for (int mt = 0; mt < 4; ++mt) {
            Frag A;
            A.u[0] = *(const uint4v*)(cur + ((mt * 16 + ln) * 512 + kb));
            A.u[1] = *(const uint4v*)(cur + ((mt * 16 + ln) * 512 + kb + 16));
            acc[0][mt] = __builtin_amdgcn_wmma_f32_16x16x32_bf16(
                    false, A.bf, false, B0.bf, (short)0, acc[0][mt], false, false);
            acc[1][mt] = __builtin_amdgcn_wmma_f32_16x16x32_bf16(
                    false, A.bf, false, B1.bf, (short)0, acc[1][mt], false, false);
        }
    }
}

// ---------------------------------------------------------------------------
// Encoder: fused mu-MLP + logvar-MLP + reparameterize.
// One block = 64 rows.  4 waves split the N-tiles (2 per wave per pass).
// Dynamic LDS: actA(64KB) + actB(64KB) + muStage(16KB) = 144KB.
// ---------------------------------------------------------------------------
struct EncArgs {
    Layer L[6];                 // 0..2 mu, 3..5 logvar (all leaky)
    const unsigned short* X;    // rows x 64 bf16
    const float* eps;           // rows x 64 f32
    float* z;                   // rows x 64 f32
};

__global__ __launch_bounds__(32 * NWAVE) void encoder_fused(EncArgs a) {
    extern __shared__ char smem[];
    unsigned short* actA = (unsigned short*)smem;
    unsigned short* actB = actA + MROWS * 512;
    float* stage = (float*)(smem + 2 * MROWS * 512 * 2);

    const int tid  = threadIdx.x;
    const int wave = tid >> 5;
    const int lane = tid & 31;
    const int hi = lane >> 4;
    const int ln = lane & 15;
    const long row0 = (long)blockIdx.x * MROWS;

    for (int pass = 0; pass < 2; ++pass) {
        load_tile(a.X, row0, 64, actA, tid);
        __syncthreads();

        unsigned short* cur = actA;
        unsigned short* nxt = actB;

        for (int l = 0; l < 3; ++l) {
            const Layer L = a.L[pass * 3 + l];
            const int kt_n = L.din_pad >> 5;
            const int nt_n = L.dout_pad >> 4;
            const bool lastL = (l == 2);

            for (int nt0 = wave * 2; nt0 < nt_n; nt0 += 2 * NWAVE) {
                const int nvalid = (nt0 + 1 < nt_n) ? 2 : 1;
                const int nt1 = (nvalid == 2) ? nt0 + 1 : nt0;   // clamp
                const int col0 = nt0 * 16 + ln;
                const int col1 = nt1 * 16 + ln;
                const unsigned short* wrow0 = L.Wt + (long)col0 * L.din_pad;
                const unsigned short* wrow1 = L.Wt + (long)col1 * L.din_pad;

                v8f acc[2][4];
#pragma unroll
                for (int j = 0; j < 2; ++j)
#pragma unroll
                    for (int mt = 0; mt < 4; ++mt) acc[j][mt] = (v8f){};

                gemm_patch(cur, wrow0, wrow1, kt_n, hi, ln, acc);

                const float bv0 = L.bias[col0];
                const float bv1 = L.bias[col1];
#pragma unroll
                for (int j = 0; j < 2; ++j) {
                    const float bv = j ? bv1 : bv0;
#pragma unroll
                    for (int mt = 0; mt < 4; ++mt)
#pragma unroll
                        for (int r = 0; r < 8; ++r) {
                            float v = acc[j][mt][r] + bv;
                            acc[j][mt][r] = v > 0.f ? v : 0.01f * v;  // all leaky
                        }
                }

                for (int j = 0; j < nvalid; ++j) {
                    const int col = j ? col1 : col0;
                    if (!lastL) {
#pragma unroll
                        for (int mt = 0; mt < 4; ++mt)
#pragma unroll
                            for (int r = 0; r < 8; ++r)
                                nxt[(mt * 16 + r + hi * 8) * 512 + col] = f2bf(acc[j][mt][r]);
                    } else if (pass == 0) {
                        // stage mu tile (re-read by the same lanes in pass 1)
#pragma unroll
                        for (int mt = 0; mt < 4; ++mt)
#pragma unroll
                            for (int r = 0; r < 8; ++r)
                                stage[(mt * 16 + r + hi * 8) * 64 + col] = acc[j][mt][r];
                    } else {
                        // z = eps * exp(0.5*logvar) + mu
#pragma unroll
                        for (int mt = 0; mt < 4; ++mt)
#pragma unroll
                            for (int r = 0; r < 8; ++r) {
                                int m = mt * 16 + r + hi * 8;
                                long row = row0 + m;
                                float mu = stage[m * 64 + col];
                                float lv = acc[j][mt][r];
                                a.z[row * 64 + col] =
                                    a.eps[row * 64 + col] * __expf(0.5f * lv) + mu;
                            }
                    }
                }
            }
            __syncthreads();
            unsigned short* t = cur; cur = nxt; nxt = t;
        }
    }
}

// ---------------------------------------------------------------------------
// Decoder: 5-layer fused MLP, same blocking.  Dynamic LDS: 128KB.
// ---------------------------------------------------------------------------
struct DecArgs {
    Layer L[5];
    int nl;
    const unsigned short* X;    // rows x ldx bf16
    int ldx;
    float* Out;                 // rows x ldo f32
    int ldo;
};

__global__ __launch_bounds__(32 * NWAVE) void decoder_fused(DecArgs a) {
    extern __shared__ char smem[];
    unsigned short* actA = (unsigned short*)smem;
    unsigned short* actB = actA + MROWS * 512;

    const int tid  = threadIdx.x;
    const int wave = tid >> 5;
    const int lane = tid & 31;
    const int hi = lane >> 4;
    const int ln = lane & 15;
    const long row0 = (long)blockIdx.x * MROWS;

    load_tile(a.X, row0, a.ldx, actA, tid);
    __syncthreads();

    unsigned short* cur = actA;
    unsigned short* nxt = actB;

    for (int l = 0; l < a.nl; ++l) {
        const Layer L = a.L[l];
        const int kt_n = L.din_pad >> 5;
        const int nt_n = L.dout_pad >> 4;
        const bool lastL = (l == a.nl - 1);

        for (int nt0 = wave * 2; nt0 < nt_n; nt0 += 2 * NWAVE) {
            const int nvalid = (nt0 + 1 < nt_n) ? 2 : 1;
            const int nt1 = (nvalid == 2) ? nt0 + 1 : nt0;   // clamp
            const int col0 = nt0 * 16 + ln;
            const int col1 = nt1 * 16 + ln;
            const unsigned short* wrow0 = L.Wt + (long)col0 * L.din_pad;
            const unsigned short* wrow1 = L.Wt + (long)col1 * L.din_pad;

            v8f acc[2][4];
#pragma unroll
            for (int j = 0; j < 2; ++j)
#pragma unroll
                for (int mt = 0; mt < 4; ++mt) acc[j][mt] = (v8f){};

            gemm_patch(cur, wrow0, wrow1, kt_n, hi, ln, acc);

            const float bv0 = L.bias[col0];
            const float bv1 = L.bias[col1];
#pragma unroll
            for (int j = 0; j < 2; ++j) {
                const float bv = j ? bv1 : bv0;
#pragma unroll
                for (int mt = 0; mt < 4; ++mt)
#pragma unroll
                    for (int r = 0; r < 8; ++r) {
                        float v = acc[j][mt][r] + bv;
                        if (L.act) v = v > 0.f ? v : 0.01f * v;
                        acc[j][mt][r] = v;
                    }
            }

            for (int j = 0; j < nvalid; ++j) {
                const int col = j ? col1 : col0;
                if (lastL) {
                    if (col < L.dout) {
#pragma unroll
                        for (int mt = 0; mt < 4; ++mt)
#pragma unroll
                            for (int r = 0; r < 8; ++r) {
                                int m = mt * 16 + r + hi * 8;
                                a.Out[(row0 + m) * a.ldo + col] = acc[j][mt][r];
                            }
                    }
                } else {
#pragma unroll
                    for (int mt = 0; mt < 4; ++mt)
#pragma unroll
                        for (int r = 0; r < 8; ++r)
                            nxt[(mt * 16 + r + hi * 8) * 512 + col] = f2bf(acc[j][mt][r]);
                }
            }
        }
        __syncthreads();
        unsigned short* t = cur; cur = nxt; nxt = t;
    }
}

// ---------------------------------------------------------------------------
// Fused 6-layer GRU stack, thread-per-batch.  All state in VGPRs via full
// unrolling; weights are wave-uniform -> scalar loads.
// ---------------------------------------------------------------------------
struct GruP { const float* Wih; const float* Whh; const float* bih; const float* bhh; };
struct GruArgs { GruP g[6]; };

template <int DIN>
__device__ __forceinline__ void gru_cell(const GruP& g, const float* __restrict__ inp,
                                         float h[HH]) {
    float gi[3 * HH], gh[3 * HH];
#pragma unroll
    for (int q = 0; q < 3 * HH; ++q) {
        float s = g.bih[q];
#pragma unroll
        for (int k = 0; k < DIN; ++k) s += g.Wih[q * DIN + k] * inp[k];
        gi[q] = s;
    }
#pragma unroll
    for (int q = 0; q < 3 * HH; ++q) {
        float s = g.bhh[q];
#pragma unroll
        for (int k = 0; k < HH; ++k) s += g.Whh[q * HH + k] * h[k];
        gh[q] = s;
    }
#pragma unroll
    for (int i = 0; i < HH; ++i) {
        float r = 1.f / (1.f + __expf(-(gi[i] + gh[i])));
        float zt = 1.f / (1.f + __expf(-(gi[HH + i] + gh[HH + i])));
        float n = tanhf(gi[2 * HH + i] + r * gh[2 * HH + i]);
        h[i] = (1.f - zt) * n + zt * h[i];
    }
}

__global__ void gru_stack(const float* __restrict__ z, GruArgs a,
                          float* __restrict__ rnnLast) {
    int b = blockIdx.x * blockDim.x + threadIdx.x;
    if (b >= BB) return;

    float h[6][HH];
#pragma unroll
    for (int l = 0; l < 6; ++l)
#pragma unroll
        for (int i = 0; i < HH; ++i) h[l][i] = 0.f;

    for (int t = 0; t < TT; ++t) {
        const float* xr = z + ((long)b * TT + t) * LAT;
        float xin[LAT];
#pragma unroll
        for (int k = 0; k < LAT; ++k) xin[k] = xr[k];

        gru_cell<LAT>(a.g[0], xin, h[0]);
#pragma unroll
        for (int l = 1; l < 6; ++l) gru_cell<HH>(a.g[l], h[l - 1], h[l]);
    }
#pragma unroll
    for (int i = 0; i < HH; ++i) rnnLast[b * HH + i] = h[5][i];
}

// ---------------------------------------------------------------------------
// Host-side orchestration
// ---------------------------------------------------------------------------
extern "C" void kernel_launch(void* const* d_in, const int* in_sizes, int n_in,
                              void* d_out, int out_size, void* d_ws, size_t ws_size,
                              hipStream_t stream) {
    (void)in_sizes; (void)n_in; (void)out_size; (void)ws_size;

    const float* lower = (const float*)d_in[0];   // (B,T,27)
    const float* hands = (const float*)d_in[1];   // (B,18)
    const float* eps   = (const float*)d_in[2];   // (B,T,64)
    // enc_mu (W,b)x3 -> 3..8 ; enc_lv -> 9..14 ; gru 6x4 -> 15..38 ;
    // dec_in 39,40 ; dec_hidden 41..44 ; final 45..48

    char* ws = (char*)d_ws;
    size_t off = 0;
    auto alloc = [&](size_t bytes) -> void* {
        void* p = ws + off;
        off = (off + bytes + 255) & ~(size_t)255;
        return p;
    };

    unsigned short* Xenc = (unsigned short*)alloc((size_t)NROWS * 64 * 2);
    float* zBuf = (float*)alloc((size_t)NROWS * 64 * 4);
    float* rnnLast = (float*)alloc((size_t)BB * HH * 4);
    unsigned short* Xdec = (unsigned short*)alloc((size_t)BB * 32 * 2);

    struct WG { int wi, bi, din, dout, dinp, doutp; };
    const WG wg[11] = {
        {3, 4, 45, 512, 64, 512},  {5, 6, 512, 512, 512, 512},  {7, 8, 512, 64, 512, 64},     // enc_mu
        {9, 10, 45, 512, 64, 512}, {11, 12, 512, 512, 512, 512},{13, 14, 512, 64, 512, 64},   // enc_lv
        {39, 40, 26, 512, 32, 512},{41, 42, 512, 512, 512, 512},{43, 44, 512, 512, 512, 512}, // dec
        {45, 46, 512, 512, 512, 512},{47, 48, 512, 27, 512, 32}                               // final
    };
    unsigned short* Wt[11];
    float* Bp[11];
    for (int i = 0; i < 11; ++i) {
        Wt[i] = (unsigned short*)alloc((size_t)wg[i].dinp * wg[i].doutp * 2);
        Bp[i] = (float*)alloc((size_t)wg[i].doutp * 4);
        int total = wg[i].dinp * wg[i].doutp;
        prep_weight<<<(total + 255) / 256, 256, 0, stream>>>(
            (const float*)d_in[wg[i].wi], (const float*)d_in[wg[i].bi],
            Wt[i], Bp[i], wg[i].din, wg[i].dout, wg[i].dinp, wg[i].doutp);
    }

    // Encoder input (padded bf16)
    {
        long total = (long)NROWS * 64;
        build_enc_input<<<(unsigned)((total + 255) / 256), 256, 0, stream>>>(lower, hands, Xenc);
    }

    // Encoder (mu + logvar + reparameterize) — 144KB dynamic LDS per block
    {
        const int encSmem = 2 * MROWS * 512 * 2 + MROWS * 64 * 4;
        (void)hipFuncSetAttribute((const void*)encoder_fused,
                                  hipFuncAttributeMaxDynamicSharedMemorySize, encSmem);
        EncArgs a{};
        for (int l = 0; l < 6; ++l)
            a.L[l] = Layer{Wt[l], Bp[l], wg[l].dinp, wg[l].doutp, wg[l].dout, 1};
        a.X = Xenc; a.eps = eps; a.z = zBuf;
        encoder_fused<<<NROWS / MROWS, 32 * NWAVE, encSmem, stream>>>(a);
    }

    // Fused GRU stack -> rnn_last
    {
        GruArgs g{};
        for (int l = 0; l < 6; ++l) {
            int base = 15 + 4 * l;
            g.g[l] = GruP{(const float*)d_in[base + 0], (const float*)d_in[base + 1],
                          (const float*)d_in[base + 2], (const float*)d_in[base + 3]};
        }
        gru_stack<<<BB / 256, 256, 0, stream>>>(zBuf, g, rnnLast);
    }

    // Decoder input (padded bf16)
    build_dec_input<<<(BB * 32 + 255) / 256, 256, 0, stream>>>(rnnLast, hands, Xdec);

    // Decoder: 26->512 (none), 3x 512->512 (leaky), 512->27 (none)
    {
        const int decSmem = 2 * MROWS * 512 * 2;
        (void)hipFuncSetAttribute((const void*)decoder_fused,
                                  hipFuncAttributeMaxDynamicSharedMemorySize, decSmem);
        DecArgs a{};
        const int acts[5] = {0, 1, 1, 1, 0};
        for (int l = 0; l < 5; ++l)
            a.L[l] = Layer{Wt[6 + l], Bp[6 + l], wg[6 + l].dinp,
                           wg[6 + l].doutp, wg[6 + l].dout, acts[l]};
        a.nl = 5; a.X = Xdec; a.ldx = 32; a.Out = (float*)d_out; a.ldo = 27;
        decoder_fused<<<BB / MROWS, 32 * NWAVE, decSmem, stream>>>(a);
    }
}
